// GCN_15092515078265
// MI455X (gfx1250) — matrix operand order repaired and checked by reading, not compiled, not run on hardware
//
#include <hip/hip_runtime.h>

typedef __attribute__((ext_vector_type(16))) __bf16 v16bf;
typedef __attribute__((ext_vector_type(8)))  __bf16 v8bf;
typedef __attribute__((ext_vector_type(4)))  __bf16 v4bf;
typedef __attribute__((ext_vector_type(8)))  float  v8f;
typedef __attribute__((ext_vector_type(4)))  unsigned int u32x4;
typedef __attribute__((ext_vector_type(8)))  int      i32x8;
typedef __attribute__((ext_vector_type(4)))  int      i32x4;

#define N_NODES 50000
#define N_EDGES 800000
#define N_REL   4
#define N_BASES 30
#define G_DIM   96
#define H1_DIM  128
#define H2_DIM  128
#define NPAD    50048   /* 391 * 128 : M padded to multiple of 128 */

/* ---------------- workspace layout (bytes; all 256-aligned) ----------------
   sums  f32 [N*R*G]        @ 0           76,800,000
   cnt   f32 [N*R]          @ 76,800,000     800,000
   agg   f32 [NPAD*128]     @ 77,600,000  25,624,576
   Acat  bf16 [NPAD*480]    @ 103,224,576 48,046,080
   Bcat  bf16 [NPAD*256]    @ 151,270,656 25,624,576   (cols 0-127 = agg, 128-255 = h)
   Wt1   bf16 [128][480]    @ 176,895,232    122,880   (K-contiguous, transposed)
   Wt2   bf16 [128][256]    @ 177,018,112     65,536
   total 177,083,648 bytes
--------------------------------------------------------------------------- */

__global__ void k_zero_f4(float4* __restrict__ p, long long n4) {
  long long t = (long long)blockIdx.x * blockDim.x + threadIdx.x;
  if (t < n4) p[t] = make_float4(0.f, 0.f, 0.f, 0.f);
}

/* Build transposed bf16 weight matrices:
   Wt1[o][k], k<384: W[r=k/96][i=k%96][o] = sum_b comp[r,b]*basis[b,i,o]
              k>=384: root[k-384][o]
   Wt2[o][k], k<128: w_rel[k][o];  k>=128: w_root[k-128][o]                  */
__global__ void k_weights(const float* __restrict__ comp,
                          const float* __restrict__ basis,
                          const float* __restrict__ root,
                          const float* __restrict__ w_rel,
                          const float* __restrict__ w_root,
                          __bf16* __restrict__ Wt1,
                          __bf16* __restrict__ Wt2) {
  int t = blockIdx.x * blockDim.x + threadIdx.x;
  const int T1 = 128 * 480;
  const int T2 = 128 * 256;
  if (t < T1) {
    int o = t / 480, kk = t - o * 480;
    float v;
    if (kk < 384) {
      int r = kk / 96, i = kk - r * 96;
      v = 0.f;
      #pragma unroll
      for (int b = 0; b < N_BASES; ++b)
        v += comp[r * N_BASES + b] * basis[((long long)b * 96 + i) * 128 + o];
    } else {
      v = root[(long long)(kk - 384) * 128 + o];
    }
    Wt1[t] = (__bf16)v;
  } else if (t < T1 + T2) {
    int t2 = t - T1;
    int o = t2 / 256, kk = t2 - o * 256;
    float v = (kk < 128) ? w_rel[(long long)kk * 128 + o]
                         : w_root[(long long)(kk - 128) * 128 + o];
    Wt2[t2] = (__bf16)v;
  }
}

/* Edge scatter: sums[(dst*R+etype)*96 + g] += x[src][g]; cnt[dst*R+etype] += 1. */
__global__ void k_scatter_msg(const float* __restrict__ nf,
                              const int* __restrict__ src,
                              const int* __restrict__ dst,
                              const int* __restrict__ etype,
                              float* __restrict__ sums,
                              float* __restrict__ cnt) {
  long long t = (long long)blockIdx.x * blockDim.x + threadIdx.x;
  if (t >= (long long)N_EDGES * 24) return;
  int e = (int)(t / 24), q = (int)(t - (long long)e * 24);
  int s = src[e], d = dst[e], r = etype[e];
  float4 v = *(const float4*)(nf + (long long)s * G_DIM + q * 4);
  float* base = sums + ((long long)d * N_REL + r) * G_DIM + q * 4;
  atomicAdd(base + 0, v.x);
  atomicAdd(base + 1, v.y);
  atomicAdd(base + 2, v.z);
  atomicAdd(base + 3, v.w);
  if (q == 0) atomicAdd(cnt + (long long)d * N_REL + r, 1.0f);
}

/* A = [ mean(n,r,g) r-major | node_features ] as bf16, rows >= N zeroed. */
__global__ void k_build_A(const float* __restrict__ nf,
                          const float* __restrict__ sums,
                          const float* __restrict__ cnt,
                          __bf16* __restrict__ Acat) {
  long long t = (long long)blockIdx.x * blockDim.x + threadIdx.x;
  if (t >= (long long)NPAD * 480) return;
  int n = (int)(t / 480), c = (int)(t - (long long)n * 480);
  float v = 0.f;
  if (n < N_NODES) {
    if (c < 384) {
      int r = c / 96, g = c - r * 96;
      float cc = cnt[(long long)n * N_REL + r];
      v = sums[((long long)n * N_REL + r) * G_DIM + g] / fmaxf(cc, 1.0f);
    } else {
      v = nf[(long long)n * G_DIM + (c - 384)];
    }
  }
  Acat[t] = (__bf16)v;
}

/* agg[dst] += h[src]; h read as bf16 from Bcat cols 128..255. */
__global__ void k_scatter_h(const __bf16* __restrict__ Bcat,
                            const int* __restrict__ src,
                            const int* __restrict__ dst,
                            float* __restrict__ agg) {
  long long t = (long long)blockIdx.x * blockDim.x + threadIdx.x;
  if (t >= (long long)N_EDGES * 32) return;
  int e = (int)(t / 32), q = (int)(t - (long long)e * 32);
  int s = src[e], d = dst[e];
  v4bf h4 = *(const v4bf*)(Bcat + (long long)s * 256 + 128 + q * 4);
  float* ap = agg + (long long)d * H1_DIM + q * 4;
  atomicAdd(ap + 0, (float)h4[0]);
  atomicAdd(ap + 1, (float)h4[1]);
  atomicAdd(ap + 2, (float)h4[2]);
  atomicAdd(ap + 3, (float)h4[3]);
}

/* Fill Bcat cols 0..127 with bf16(agg). */
__global__ void k_build_B(const float* __restrict__ agg,
                          __bf16* __restrict__ Bcat) {
  long long t = (long long)blockIdx.x * blockDim.x + threadIdx.x;
  if (t >= (long long)NPAD * 128) return;
  int n = (int)(t / 128), c = (int)(t - (long long)n * 128);
  Bcat[(long long)n * 256 + c] = (__bf16)agg[t];
}

/* ---------------- WMMA GEMM: C[rows x 128] = A[rows x K] * Wt^T + bias ------
   Block = 256 threads = 8 wave32; wave w owns rows [blk*128 + w*16, +16).
   Wt (128 x K bf16, K-contiguous, <=123 KB) is staged into LDS once per block:
   preferred path = Tensor Data Mover (tensor_load_to_lds + s_wait_tensorcnt),
   issued by wave 0 only (TDM ignores EXEC, so whole waves branch around it);
   fallback = cooperative uint4 copy (ds_store_b128).
   Inner loop: A-frags via 2x16B global loads per lane (prefetched ahead);
   B-frags via 2x ds_load_b128 per tile -> WMMA waits are DScnt, not L2.      */
__global__ __launch_bounds__(256)
void k_gemm_wmma(const __bf16* __restrict__ A, int K,
                 const __bf16* __restrict__ Wt,
                 const float* __restrict__ bias,
                 float* __restrict__ outF,
                 __bf16* __restrict__ outB,
                 int ldOut, int colOff, int nValid) {
  extern __shared__ __bf16 sW[];              /* [128][K] */
  const int tid  = threadIdx.x;
  const int wave = tid >> 5;
  const int lane = tid & 31;
  const int half = lane >> 4;
  const int l15  = lane & 15;
  const long long m0 = (long long)blockIdx.x * 128 + wave * 16;

  /* ---- stage Wt (256*K bytes) into LDS ---- */
#if defined(__has_builtin) && __has_builtin(__builtin_amdgcn_tensor_load_to_lds)
  if (wave == 0) {
    unsigned long long ga = (unsigned long long)(uintptr_t)Wt;
    unsigned ldsOff = (unsigned)(uintptr_t)&sW[0];   /* low 32b of LDS aperture addr */
    int nE = K * 32;                                 /* 8-byte elements to move */
    u32x4 g0;
    g0[0] = 1u;                                      /* count=1, user descriptor */
    g0[1] = ldsOff;                                  /* lds_addr */
    g0[2] = (unsigned)(ga & 0xFFFFFFFFull);          /* global_addr[31:0] */
    g0[3] = (unsigned)((ga >> 32) & 0x1FFFFFFull) | (2u << 30); /* addr[56:32] | type=2 */
    i32x8 g1;
    g1[0] = (3 << 16);                               /* data_size=8B; mask=0 */
    g1[1] = (int)((nE & 0xFFFF) << 16);              /* tensor_dim0[15:0] */
    g1[2] = (int)(((nE >> 16) & 0xFFFF) | (1 << 16)); /* tensor_dim0[31:16] | tensor_dim1=1 */
    g1[3] = (int)(nE << 16);                         /* tile_dim0 = nE (fits 16b: <=15360) */
    g1[4] = 0;                                       /* tile_dim1/2 unused */
    g1[5] = nE;                                      /* tensor_dim0_stride[31:0] */
    g1[6] = 0;
    g1[7] = 0;
    i32x4 gz = {0, 0, 0, 0};
#if __clang_major__ >= 23
    i32x8 gz8 = {0, 0, 0, 0, 0, 0, 0, 0};
    __builtin_amdgcn_tensor_load_to_lds(g0, g1, gz, gz, gz8, 0);
#else
    __builtin_amdgcn_tensor_load_to_lds(g0, g1, gz, gz, 0);
#endif
    __builtin_amdgcn_s_wait_tensorcnt(0);
  }
#else
  {
    const uint4* gsrc = (const uint4*)Wt;
    uint4* ldst = (uint4*)&sW[0];
    for (int i = tid; i < K * 16; i += 256)          /* K*256 bytes / 16 */
      ldst[i] = gsrc[i];
  }
#endif
  __syncthreads();

  v8f acc[8];
  #pragma unroll
  for (int c = 0; c < 8; ++c) {
    float b = bias[c * 16 + l15];          /* N depends only on lane&15 */
    #pragma unroll
    for (int j = 0; j < 8; ++j) acc[c][j] = b;
  }

  const __bf16* Ap = A + (m0 + l15) * K;
  for (int k0 = 0; k0 < K; k0 += 32) {
    __builtin_prefetch(Ap + k0 + 64, 0, 1);          /* global_prefetch_b8 */
    v8bf alo = *(const v8bf*)(Ap + k0 + half * 8);
    v8bf ahi = *(const v8bf*)(Ap + k0 + 16 + half * 8);
    v16bf afrag = __builtin_shufflevector(alo, ahi,
        0, 1, 2, 3, 4, 5, 6, 7, 8, 9, 10, 11, 12, 13, 14, 15);
    #pragma unroll
    for (int c = 0; c < 8; ++c) {
      const __bf16* Wp = sW + (c * 16 + l15) * K + k0 + half * 8;
      v8bf blo = *(const v8bf*)(Wp);                 /* ds_load_b128 */
      v8bf bhi = *(const v8bf*)(Wp + 16);            /* ds_load_b128 */
      v16bf bfrag = __builtin_shufflevector(blo, bhi,
          0, 1, 2, 3, 4, 5, 6, 7, 8, 9, 10, 11, 12, 13, 14, 15);
      acc[c] = __builtin_amdgcn_wmma_f32_16x16x32_bf16(
          false, afrag, false, bfrag, (short)0, acc[c], false, false);
    }
  }

  #pragma unroll
  for (int c = 0; c < 8; ++c) {
    #pragma unroll
    for (int j = 0; j < 8; ++j) {
      long long row = m0 + half * 8 + j;   /* C/D: VGPR j -> M=j (+8 for hi lanes) */
      int col = c * 16 + l15;
      float v = acc[c][j];
      if (outF) {
        if (row < nValid) outF[row * 128 + col] = v;
      } else {
        outB[row * (long long)ldOut + colOff + col] = (__bf16)v;
      }
    }
  }
}

extern "C" void kernel_launch(void* const* d_in, const int* in_sizes, int n_in,
                              void* d_out, int out_size, void* d_ws, size_t ws_size,
                              hipStream_t stream) {
  const float* nf     = (const float*)d_in[0];
  const int*   eidx   = (const int*)d_in[1];
  /* d_in[2] = edge_norm: unused by reference */
  const int*   etype  = (const int*)d_in[3];
  const float* basis  = (const float*)d_in[4];
  const float* comp   = (const float*)d_in[5];
  const float* root   = (const float*)d_in[6];
  const float* bias1  = (const float*)d_in[7];
  const float* w_rel  = (const float*)d_in[8];
  const float* b_rel  = (const float*)d_in[9];
  const float* w_root = (const float*)d_in[10];
  const int* src = eidx;
  const int* dst = eidx + N_EDGES;

  char* ws = (char*)d_ws;
  float*  sums = (float*)(ws + 0);
  float*  cnt  = (float*)(ws + 76800000LL);
  float*  agg  = (float*)(ws + 77600000LL);
  __bf16* Acat = (__bf16*)(ws + 103224576LL);
  __bf16* Bcat = (__bf16*)(ws + 151270656LL);
  __bf16* Wt1  = (__bf16*)(ws + 176895232LL);
  __bf16* Wt2  = (__bf16*)(ws + 177018112LL);
  float* out = (float*)d_out;

  const int B = 256;

  /* 1) zero sums + cnt + agg (contiguous: 25,806,144 floats = 6,451,536 float4) */
  long long n4 = 6451536LL;
  k_zero_f4<<<dim3((unsigned)((n4 + B - 1) / B)), dim3(B), 0, stream>>>((float4*)ws, n4);

  /* 2) weights: comp x basis -> Wt1 (+root rows); Wt2 from w_rel/w_root */
  int wthreads = 128 * 480 + 128 * 256;
  k_weights<<<dim3((wthreads + B - 1) / B), dim3(B), 0, stream>>>(
      comp, basis, root, w_rel, w_root, Wt1, Wt2);

  /* 3) per-(node,relation) message scatter */
  long long st1 = (long long)N_EDGES * 24;
  k_scatter_msg<<<dim3((unsigned)((st1 + B - 1) / B)), dim3(B), 0, stream>>>(
      nf, src, dst, etype, sums, cnt);

  /* 4) A = [mean | x] in bf16 */
  long long at = (long long)NPAD * 480;
  k_build_A<<<dim3((unsigned)((at + B - 1) / B)), dim3(B), 0, stream>>>(
      nf, sums, cnt, Acat);

  /* 5) GEMM1: h = A @ Wt1^T + bias1  ->  bf16 into Bcat cols 128..255 */
  k_gemm_wmma<<<dim3(NPAD / 128), dim3(B), 480 * 256, stream>>>(
      Acat, 480, Wt1, bias1, nullptr, Bcat, 256, 128, NPAD);

  /* 6) agg[dst] += h[src] */
  long long st2 = (long long)N_EDGES * 32;
  k_scatter_h<<<dim3((unsigned)((st2 + B - 1) / B)), dim3(B), 0, stream>>>(
      Bcat, src, dst, agg);

  /* 7) Bcat cols 0..127 = bf16(agg) */
  long long bt = (long long)NPAD * 128;
  k_build_B<<<dim3((unsigned)((bt + B - 1) / B)), dim3(B), 0, stream>>>(agg, Bcat);

  /* 8) GEMM2: out = [agg|h] @ Wt2^T + b_rel  -> f32 d_out (guard n < N) */
  k_gemm_wmma<<<dim3(NPAD / 128), dim3(B), 256 * 256, stream>>>(
      Bcat, 256, Wt2, b_rel, out, nullptr, 128, 0, N_NODES);

  (void)in_sizes; (void)n_in; (void)out_size; (void)ws_size;
}
